// pointer_layer_41618233098735
// MI455X (gfx1250) — matrix operand order repaired and checked by reading, not compiled, not run on hardware
//
#include <hip/hip_runtime.h>
#include <hip/hip_bf16.h>

// ---------------------------------------------------------------------------
// Point-Transformer layer for MI455X (gfx1250), bf16 WMMA path.
// B=16, NQ=1024, NK=4096, D=256, K=16 neighbors.
// ---------------------------------------------------------------------------

typedef __attribute__((ext_vector_type(16))) __bf16 v16bf;
typedef __attribute__((ext_vector_type(8)))  float  v8f;

union BF16x16 { v16bf v; unsigned u[8]; };

__device__ __forceinline__ unsigned short f2bf(float f) {
    unsigned u = __float_as_uint(f);
    u = (u + 0x7fffu + ((u >> 16) & 1u)) >> 16;   // round-to-nearest-even
    return (unsigned short)u;
}
__device__ __forceinline__ unsigned f2bf2(float lo, float hi) {
    return (unsigned)f2bf(lo) | ((unsigned)f2bf(hi) << 16);
}

// ---------------------------------------------------------------------------
// Pack a [Ksrc,256] f32 weight (x @ W layout) into bf16 B-fragment order:
// dst[((kt*256 + n)*2 + h)*16 + vg*2 + s]  <-  W[(kt*32 + h*16 + vg*2 + s), n]
// (contiguous K order chosen here for A and B: lanes 0-15 hold K=0..15, lanes 16-31 hold K=16..31,
//  two K values per VGPR.)  Rows >= Ksrc are zero (K padding).
// ---------------------------------------------------------------------------
__global__ void pack_b_kernel(const float* __restrict__ W,
                              unsigned short* __restrict__ dst,
                              int Ksrc, int total) {
    int e = blockIdx.x * 256 + threadIdx.x;
    if (e >= total) return;
    int s  = e & 1;
    int vg = (e >> 1) & 7;
    int h  = (e >> 4) & 1;
    int n  = (e >> 5) & 255;
    int kt = e >> 13;
    int k  = kt * 32 + h * 16 + vg * 2 + s;
    float v = (k < Ksrc) ? W[(size_t)k * 256 + n] : 0.0f;
    dst[e] = f2bf(v);
}

// ---------------------------------------------------------------------------
// Flat GEMM: Y[M,256] = act(A[M,lda(f32)] @ Wp(bf16 packed, Kpad x 256) + bias)
// Block = 256 threads = 8 waves; wave computes a 16x64 output tile.
// Grid  = (M/128, 4).
// ---------------------------------------------------------------------------
__global__ __launch_bounds__(256)
void gemm_wmma_kernel(const float* __restrict__ A, int lda,
                      const unsigned short* __restrict__ Wp,
                      const float* __restrict__ bias,
                      float* __restrict__ Y,
                      int Kpad, int relu) {
    const int lane = threadIdx.x & 31;
    const int wave = threadIdx.x >> 5;
    const int m0   = (blockIdx.x * 8 + wave) * 16;
    const int n0   = blockIdx.y * 64;
    const int h    = lane >> 4;
    const int ml   = lane & 15;
    const int m    = m0 + ml;

    const int baseA[8] = {0, 2, 4, 6, 8, 10, 12, 14};

    v8f acc[4] = {};
    const int ksteps = Kpad >> 5;
    const float* arow = A + (size_t)m * lda;

    for (int kt = 0; kt < ksteps; ++kt) {
        BF16x16 a;
#pragma unroll
        for (int vg = 0; vg < 8; ++vg) {
            const float2 f = *(const float2*)(arow + kt * 32 + baseA[vg] + h * 16);
            a.u[vg] = f2bf2(f.x, f.y);
        }
        if (kt + 1 < ksteps)
            __builtin_prefetch(arow + (kt + 1) * 32 + h * 8, 0, 0);
#pragma unroll
        for (int j = 0; j < 4; ++j) {
            const int n = n0 + j * 16 + ml;
            const unsigned* b32 = (const unsigned*)Wp +
                (((size_t)(kt * 256 + n) * 2 + h) * 8);
            BF16x16 bf;
            uint4 q0 = *(const uint4*)b32;
            uint4 q1 = *(const uint4*)(b32 + 4);
            bf.u[0] = q0.x; bf.u[1] = q0.y; bf.u[2] = q0.z; bf.u[3] = q0.w;
            bf.u[4] = q1.x; bf.u[5] = q1.y; bf.u[6] = q1.z; bf.u[7] = q1.w;
            acc[j] = __builtin_amdgcn_wmma_f32_16x16x32_bf16(
                false, a.v, false, bf.v, (short)0, acc[j], false, false);
        }
    }

#pragma unroll
    for (int j = 0; j < 4; ++j) {
        const int col = n0 + j * 16 + ml;
        const float bv = bias ? bias[col] : 0.0f;
#pragma unroll
        for (int r = 0; r < 8; ++r) {
            const int row = m0 + r + h * 8;
            float v = acc[j][r] + bv;
            if (relu) v = fmaxf(v, 0.0f);
            Y[(size_t)row * 256 + col] = v;
        }
    }
}

// ---------------------------------------------------------------------------
// Brute-force 16-NN: one thread per query point, register insertion sort.
// qxyz: [16*1024,3]; kxyz: [16,Mk,3]; out: [16*1024,16]
// ---------------------------------------------------------------------------
__global__ void knn_kernel(const float* __restrict__ qxyz,
                           const float* __restrict__ kxyz,
                           int Mk, int* __restrict__ out) {
    int tid = blockIdx.x * 256 + threadIdx.x;
    if (tid >= 16384) return;
    int b = tid >> 10;
    float qx = qxyz[tid * 3], qy = qxyz[tid * 3 + 1], qz = qxyz[tid * 3 + 2];
    const float* kb = kxyz + (size_t)b * Mk * 3;
    float bd[16]; int bi[16];
#pragma unroll
    for (int i = 0; i < 16; ++i) { bd[i] = 3.4e38f; bi[i] = 0; }
    float worst = 3.4e38f;
    for (int mm = 0; mm < Mk; ++mm) {
        float dx = qx - kb[mm * 3];
        float dy = qy - kb[mm * 3 + 1];
        float dz = qz - kb[mm * 3 + 2];
        float d = dx * dx + dy * dy + dz * dz;
        if (d < worst) {
            int p = 15;
            while (p > 0 && bd[p - 1] > d) { bd[p] = bd[p - 1]; bi[p] = bi[p - 1]; --p; }
            bd[p] = d; bi[p] = mm;
            worst = bd[15];
        }
    }
    for (int i = 0; i < 16; ++i) out[tid * 16 + i] = bi[i];
}

// ---------------------------------------------------------------------------
// Build padded concat(xyz, qf) -> [16384, 288] with zero K-padding.
// ---------------------------------------------------------------------------
__global__ void qcat_kernel(const float* __restrict__ qxyz,
                            const float* __restrict__ qf,
                            float* __restrict__ dst) {
    int e = blockIdx.x * 256 + threadIdx.x;
    if (e >= 16384 * 288) return;
    int row = e / 288, col = e - row * 288;
    float v;
    if (col < 3)        v = qxyz[row * 3 + col];
    else if (col < 259) v = qf[(size_t)row * 256 + (col - 3)];
    else                v = 0.0f;
    dst[e] = v;
}

// ---------------------------------------------------------------------------
// LDS-staged 16x256 = (16x256 bf16) @ (256x256 packed bf16) WMMA tile.
// 16 waves, wave w -> output columns [16w,16w+16). dst gets +bias (+ReLU).
// ---------------------------------------------------------------------------
__device__ __forceinline__
void wmma_tile(const unsigned short (&s16)[16][256],
               const unsigned short* __restrict__ Wp,
               const float* __restrict__ bias,
               float (&dst)[16][256],
               int relu, int t) {
    const int lane = t & 31;
    const int wave = t >> 5;
    const int h    = lane >> 4;
    const int ml   = lane & 15;
    const int n0   = wave * 16;
    const int baseA[8] = {0, 2, 4, 6, 8, 10, 12, 14};

    v8f acc = {};
    for (int kt = 0; kt < 8; ++kt) {
        BF16x16 a;
#pragma unroll
        for (int vg = 0; vg < 8; ++vg) {
            const int k = kt * 32 + baseA[vg] + h * 16;
            a.u[vg] = *(const unsigned*)&s16[ml][k];
        }
        const unsigned* b32 = (const unsigned*)Wp +
            (((size_t)(kt * 256 + n0 + ml) * 2 + h) * 8);
        BF16x16 bf;
        uint4 q0 = *(const uint4*)b32;
        uint4 q1 = *(const uint4*)(b32 + 4);
        bf.u[0] = q0.x; bf.u[1] = q0.y; bf.u[2] = q0.z; bf.u[3] = q0.w;
        bf.u[4] = q1.x; bf.u[5] = q1.y; bf.u[6] = q1.z; bf.u[7] = q1.w;
        acc = __builtin_amdgcn_wmma_f32_16x16x32_bf16(
            false, a.v, false, bf.v, (short)0, acc, false, false);
    }
    const int col = n0 + ml;
    const float bv = bias ? bias[col] : 0.0f;
#pragma unroll
    for (int r = 0; r < 8; ++r) {
        float v = acc[r] + bv;
        if (relu) v = fmaxf(v, 0.0f);
        dst[r + h * 8][col] = v;
    }
}

// ---------------------------------------------------------------------------
// Fused Point-Transformer vector attention, one block per query point.
// mode 0 (self): kfull/vfull are precomputed wk(x)/wv(x) rows to gather.
// mode 1 (cross): kfull = xk rows; kk/vv computed in-kernel via WMMA.
// ---------------------------------------------------------------------------
__global__ __launch_bounds__(512)
void pt_attn_kernel(int mode,
                    const float* __restrict__ q_xyz,
                    const float* __restrict__ k_xyz, int Mk,
                    const float* __restrict__ qvec,
                    const float* __restrict__ kfull,
                    const float* __restrict__ vfull,
                    const unsigned short* __restrict__ wkp,
                    const unsigned short* __restrict__ wvp,
                    const int* __restrict__ idx,
                    const float* __restrict__ d1w, const float* __restrict__ d1b,
                    const unsigned short* __restrict__ d2w, const float* __restrict__ d2b,
                    const unsigned short* __restrict__ g1w, const float* __restrict__ g1b,
                    const unsigned short* __restrict__ g2w, const float* __restrict__ g2b,
                    const float* __restrict__ fc2w, const float* __restrict__ fc2b,
                    const float* __restrict__ pre,
                    float* __restrict__ out) {
    __shared__ float bufB[16][256];              // pos
    __shared__ float bufC[16][256];              // vv
    __shared__ float bufD[16][256];              // kk -> gamma hidden -> logits
    __shared__ unsigned short s16[16][256];      // bf16 WMMA A staging
    __shared__ float qrow[256];
    __shared__ float resv[256];
    __shared__ float diffs[16][3];
    __shared__ int   sidx[16];
    __shared__ float qpt[3];

    const int bn = blockIdx.x;
    const int b  = bn >> 10;                     // NQ = 1024
    const int t  = threadIdx.x;

    if (t < 16)               sidx[t]       = idx[bn * 16 + t];
    if (t >= 32 && t < 35)    qpt[t - 32]   = q_xyz[bn * 3 + (t - 32)];
    if (t >= 64 && t < 320)   qrow[t - 64]  = qvec[(size_t)bn * 256 + (t - 64)];
    __syncthreads();

    if (t < 16) {
        const float* kp = k_xyz + ((size_t)b * Mk + sidx[t]) * 3;
        diffs[t][0] = qpt[0] - kp[0];
        diffs[t][1] = qpt[1] - kp[1];
        diffs[t][2] = qpt[2] - kp[2];
    }
    {   // gather neighbor rows (8 channels per thread)
        const int j  = t >> 5;
        const int c0 = (t & 31) * 8;
        const size_t rb = ((size_t)b * Mk + sidx[j]) * 256 + c0;
        if (mode == 0) {
#pragma unroll
            for (int i = 0; i < 8; ++i) bufD[j][c0 + i] = kfull[rb + i];
#pragma unroll
            for (int i = 0; i < 8; ++i) bufC[j][c0 + i] = vfull[rb + i];
        } else {
#pragma unroll
            for (int i = 0; i < 8; ++i) s16[j][c0 + i] = f2bf(kfull[rb + i]);
        }
    }
    __syncthreads();

    if (mode == 1) {                             // kk = xk@wk, vv = xk@wv
        wmma_tile(s16, wkp, nullptr, bufD, 0, t);
        wmma_tile(s16, wvp, nullptr, bufC, 0, t);
        __syncthreads();
    }

    {   // delta layer 1: h = relu(diff @ W1 + b1) -> s16 (K=3, VALU)
        const int j  = t >> 5;
        const int c0 = (t & 31) * 8;
        const float dx = diffs[j][0], dy = diffs[j][1], dz = diffs[j][2];
#pragma unroll
        for (int i = 0; i < 8; ++i) {
            const int c = c0 + i;
            float a = d1b[c] + dx * d1w[c] + dy * d1w[256 + c] + dz * d1w[512 + c];
            s16[j][c] = f2bf(fmaxf(a, 0.0f));
        }
    }
    __syncthreads();
    wmma_tile(s16, d2w, d2b, bufB, 0, t);        // pos = h @ W2 + b2
    __syncthreads();

    {   // s = q - kk + pos
        const int j  = t >> 5;
        const int c0 = (t & 31) * 8;
#pragma unroll
        for (int i = 0; i < 8; ++i) {
            const int c = c0 + i;
            s16[j][c] = f2bf(qrow[c] - bufD[j][c] + bufB[j][c]);
        }
    }
    __syncthreads();
    wmma_tile(s16, g1w, g1b, bufD, 1, t);        // gamma hidden (ReLU)
    __syncthreads();
    {
        const int j  = t >> 5;
        const int c0 = (t & 31) * 8;
#pragma unroll
        for (int i = 0; i < 8; ++i) s16[j][c0 + i] = f2bf(bufD[j][c0 + i]);
    }
    __syncthreads();
    wmma_tile(s16, g2w, g2b, bufD, 0, t);        // logits
    __syncthreads();

    if (t < 256) {                               // softmax over neighbors, weighted sum
        float mx = -3.4e38f;
#pragma unroll
        for (int j = 0; j < 16; ++j) mx = fmaxf(mx, bufD[j][t]);
        float ssum = 0.0f;
#pragma unroll
        for (int j = 0; j < 16; ++j) {
            float e = __expf((bufD[j][t] - mx) * 0.0625f);   // /sqrt(256)
            bufD[j][t] = e;
            ssum += e;
        }
        const float inv = 1.0f / ssum;
        float r = 0.0f;
#pragma unroll
        for (int j = 0; j < 16; ++j) r += bufD[j][t] * (bufC[j][t] + bufB[j][t]);
        resv[t] = r * inv;
    }
    __syncthreads();

    if (t < 256) {                               // fc2 GEMV + bias + residual
        float acc = fc2b[t] + pre[(size_t)bn * 256 + t];
        for (int k = 0; k < 256; ++k) acc += resv[k] * fc2w[(size_t)k * 256 + t];
        out[(size_t)bn * 256 + t] = acc;
    }
}

// ---------------------------------------------------------------------------
// Final reg layer: delta3 = hidden @ W[256,3] + b; out = delta3 + query_xyz.
// ---------------------------------------------------------------------------
__global__ void reg_final_kernel(const float* __restrict__ hidden,
                                 const float* __restrict__ W,
                                 const float* __restrict__ bvec,
                                 const float* __restrict__ qxyz,
                                 float* __restrict__ out) {
    int tid = blockIdx.x * 256 + threadIdx.x;
    if (tid >= 16384) return;
    float a0 = bvec[0], a1 = bvec[1], a2 = bvec[2];
    const float* hr = hidden + (size_t)tid * 256;
    for (int k = 0; k < 256; ++k) {
        float hv = hr[k];
        a0 += hv * W[k * 3];
        a1 += hv * W[k * 3 + 1];
        a2 += hv * W[k * 3 + 2];
    }
    out[tid * 3]     = a0 + qxyz[tid * 3];
    out[tid * 3 + 1] = a1 + qxyz[tid * 3 + 1];
    out[tid * 3 + 2] = a2 + qxyz[tid * 3 + 2];
}

// ---------------------------------------------------------------------------
// Host-side orchestration.
// d_in flatten order (jax pytree, dicts sorted by key):
//  0 query_xyz  1 query_feats  2 key_xyz  3 key_feats
//  cross: 4 d0.b 5 d0.w 6 d1.b 7 d1.w 8 fc1k.b 9 fc1k.w 10 fc1q.b 11 fc1q.w
//         12 fc2.b 13 fc2.w 14 g0.b 15 g0.w 16 g1.b 17 g1.w 18 wk 19 wq 20 wv
//  reg:   21 r0.b 22 r0.w 23 r1.b 24 r1.w
//  self:  25 d0.b 26 d0.w 27 d1.b 28 d1.w 29 fc1q.b 30 fc1q.w 31 fc2.b 32 fc2.w
//         33 g0.b 34 g0.w 35 g1.b 36 g1.w 37 wk 38 wq 39 wv
// ---------------------------------------------------------------------------
extern "C" void kernel_launch(void* const* d_in, const int* in_sizes, int n_in,
                              void* d_out, int out_size, void* d_ws, size_t ws_size,
                              hipStream_t stream) {
    (void)in_sizes; (void)n_in; (void)out_size; (void)ws_size;

    const float* qxyz  = (const float*)d_in[0];
    const float* qfeat = (const float*)d_in[1];
    const float* kxyz  = (const float*)d_in[2];
    const float* kfeat = (const float*)d_in[3];

    const size_t MQ = 16384, MK = 65536;
    float* F = (float*)d_ws;
    const size_t fA = 0;                       // [16384,288] xq_self / qcat_pad
    const size_t fB = fA + MQ * 288;           // q_self / xq2
    const size_t fC = fB + MQ * 256;           // k_self / q_cross
    const size_t fD = fC + MQ * 256;           // v_self / reg hidden
    const size_t fF = fD + MQ * 256;           // qf (self-attn output)
    const size_t fG = fF + MQ * 256;           // xk_cross [65536,256]
    const size_t fI = fG + MK * 256;           // knn idx (ints)
    int* idxbuf = (int*)(F + fI);
    unsigned short* Wb = (unsigned short*)(F + fI + MQ * 16);
    auto slot = [&](int s) { return Wb + (size_t)s * 65536; };

    // --- pack all B-operand weights to bf16 fragment order -----------------
    const int packs[16][4] = {
        {30, 0, 256, 256}, {38, 1, 256, 256}, {37, 2, 256, 256}, {39, 3, 256, 256},
        {34, 4, 256, 256}, {36, 5, 256, 256}, {28, 6, 256, 256},
        { 9, 7, 256, 256}, {19, 8, 256, 256}, {18, 9, 256, 256}, {20,10, 256, 256},
        {15,11, 256, 256}, {17,12, 256, 256}, { 7,13, 256, 256}, {22,14, 256, 256},
        {11,15, 259, 288}};
    for (int i = 0; i < 16; ++i) {
        const int total = packs[i][3] * 256;
        pack_b_kernel<<<(total + 255) / 256, 256, 0, stream>>>(
            (const float*)d_in[packs[i][0]], slot(packs[i][1]), packs[i][2], total);
    }

    // --- self-attention pipeline ------------------------------------------
    gemm_wmma_kernel<<<dim3(128, 4), 256, 0, stream>>>(
        qfeat, 256, slot(0), (const float*)d_in[29], F + fA, 256, 0);      // xq
    gemm_wmma_kernel<<<dim3(128, 4), 256, 0, stream>>>(
        F + fA, 256, slot(1), nullptr, F + fB, 256, 0);                    // q
    gemm_wmma_kernel<<<dim3(128, 4), 256, 0, stream>>>(
        F + fA, 256, slot(2), nullptr, F + fC, 256, 0);                    // k
    gemm_wmma_kernel<<<dim3(128, 4), 256, 0, stream>>>(
        F + fA, 256, slot(3), nullptr, F + fD, 256, 0);                    // v
    knn_kernel<<<64, 256, 0, stream>>>(qxyz, qxyz, 1024, idxbuf);
    pt_attn_kernel<<<16384, 512, 0, stream>>>(
        0, qxyz, qxyz, 1024,
        F + fB, F + fC, F + fD, nullptr, nullptr, idxbuf,
        (const float*)d_in[26], (const float*)d_in[25],
        slot(6), (const float*)d_in[27],
        slot(4), (const float*)d_in[33],
        slot(5), (const float*)d_in[35],
        (const float*)d_in[32], (const float*)d_in[31],
        qfeat, F + fF);                                                    // qf

    // --- cross-attention pipeline -----------------------------------------
    gemm_wmma_kernel<<<dim3(512, 4), 256, 0, stream>>>(
        kfeat, 256, slot(7), (const float*)d_in[8], F + fG, 256, 0);       // xk
    qcat_kernel<<<(16384 * 288 + 255) / 256, 256, 0, stream>>>(
        qxyz, F + fF, F + fA);                                             // padded cat
    gemm_wmma_kernel<<<dim3(128, 4), 256, 0, stream>>>(
        F + fA, 288, slot(15), (const float*)d_in[10], F + fB, 288, 0);    // xq2
    gemm_wmma_kernel<<<dim3(128, 4), 256, 0, stream>>>(
        F + fB, 256, slot(8), nullptr, F + fC, 256, 0);                    // q_cross
    knn_kernel<<<64, 256, 0, stream>>>(qxyz, kxyz, 4096, idxbuf);
    pt_attn_kernel<<<16384, 512, 0, stream>>>(
        1, qxyz, kxyz, 4096,
        F + fC, F + fG, nullptr, slot(9), slot(10), idxbuf,
        (const float*)d_in[5], (const float*)d_in[4],
        slot(13), (const float*)d_in[6],
        slot(11), (const float*)d_in[14],
        slot(12), (const float*)d_in[16],
        (const float*)d_in[13], (const float*)d_in[12],
        F + fF, (float*)d_out);                                            // qf2

    // --- regression head ---------------------------------------------------
    gemm_wmma_kernel<<<dim3(128, 4), 256, 0, stream>>>(
        (const float*)d_out, 256, slot(14), (const float*)d_in[21],
        F + fD, 256, 1);                                                   // relu hidden
    reg_final_kernel<<<64, 256, 0, stream>>>(
        F + fD, (const float*)d_in[24], (const float*)d_in[23],
        qxyz, (float*)d_out + MQ * 256);                                   // xyz out
}
